// NDE_model_57028575756652
// MI455X (gfx1250) — matrix operand order repaired and checked by reading, not compiled, not run on hardware
//
#include <hip/hip_runtime.h>
#include <math.h>

typedef __attribute__((ext_vector_type(16))) __bf16 v16bf;
typedef __attribute__((ext_vector_type(8)))  __bf16 v8bf;
typedef __attribute__((ext_vector_type(8)))  float  v8f;
typedef unsigned int v4u __attribute__((ext_vector_type(4)));
typedef int          v8i __attribute__((ext_vector_type(8)));
typedef int          v4i __attribute__((ext_vector_type(4)));

#define Tn 2000
#define Bn 512
#define Hn 64

#if defined(__HIP_DEVICE_COMPILE__) && __has_builtin(__builtin_amdgcn_tensor_load_to_lds)
#define USE_TDM 1
#else
#define USE_TDM 0
#endif

__device__ __forceinline__ float lipswish(float x) {
  return 0.909f * x / (1.0f + __expf(-x));
}

// A-fragment (16x32 bf16, MxK) from row-major [16][64] bf16 staging buffer.
__device__ __forceinline__ v16bf load_afrag(const __bf16* bufA, int kt, int l15, int kh) {
  const __bf16* p = bufA + l15 * 64 + kt * 32 + kh * 8;
  v8bf lo = *(const v8bf*)p;
  v8bf hi = *(const v8bf*)(p + 16);
  v16bf a;
#pragma unroll
  for (int e = 0; e < 8; ++e) { a[e] = lo[e]; a[e + 8] = hi[e]; }
  return a;
}

// B-fragment from pre-swizzled LDS weight image: [kt][nt][lane][16 elems], 32B/lane.
__device__ __forceinline__ v16bf load_bfrag(const __bf16* wm, int kt, int nt, int tid) {
  const __bf16* p = wm + ((((kt << 2) | nt) * 32) + tid) * 16;
  v8bf lo = *(const v8bf*)p;
  v8bf hi = *(const v8bf*)(p + 8);
  v16bf b;
#pragma unroll
  for (int e = 0; e < 8; ++e) { b[e] = lo[e]; b[e + 8] = hi[e]; }
  return b;
}

// [16,64] x [64,64] accumulated into 4 C fragments.
__device__ __forceinline__ void gemm16x64(const __bf16* bufA, const __bf16* wm,
                                          v8f c[4], int tid, int l15, int kh) {
  v16bf a0 = load_afrag(bufA, 0, l15, kh);
  v16bf a1 = load_afrag(bufA, 1, l15, kh);
#pragma unroll
  for (int nt = 0; nt < 4; ++nt) {
    v16bf b0 = load_bfrag(wm, 0, nt, tid);
    c[nt] = __builtin_amdgcn_wmma_f32_16x16x32_bf16(false, a0, false, b0,
                                                    (short)0, c[nt], false, false);
    v16bf b1 = load_bfrag(wm, 1, nt, tid);
    c[nt] = __builtin_amdgcn_wmma_f32_16x16x32_bf16(false, a1, false, b1,
                                                    (short)0, c[nt], false, false);
  }
}

__device__ __forceinline__ void store_tile(__bf16* buf, const v8f c[4], int l15, int kh) {
#pragma unroll
  for (int nt = 0; nt < 4; ++nt)
#pragma unroll
    for (int r = 0; r < 8; ++r)
      buf[(r + 8 * kh) * 64 + nt * 16 + l15] = (__bf16)c[nt][r];
}

#if USE_TDM
// TDM: contiguous 4KB (1024 f32) global -> LDS copy. D# per ISA 08_async_tensor §8.
__device__ __forceinline__ void tdm_load_noise(const float* gsrc, unsigned lds_off) {
  unsigned long long ga = (unsigned long long)(uintptr_t)gsrc;
  v4u g0;
  g0[0] = 1u;                                   // count=1, user descriptor
  g0[1] = lds_off;                              // lds_addr
  g0[2] = (unsigned)ga;                         // global_addr[31:0]
  g0[3] = (unsigned)((ga >> 32) & 0x1FFFFFFull) // global_addr[56:32]
          | (2u << 30);                         // type=2 ("image")
  v8i g1;
  g1[0] = 0x20000;          // data_size=2 (4B), no multicast/pad/iterate
  g1[1] = 1024 << 16;       // tensor_dim0 = 1024 (low 16 bits at [63:48])
  g1[2] = 1 << 16;          // tensor_dim0 hi=0 ; tensor_dim1 = 1
  g1[3] = 1024 << 16;       // tensor_dim1 hi=0 ; tile_dim0 = 1024
  g1[4] = 1;                // tile_dim1 = 1, tile_dim2 = 0
  g1[5] = 1024;             // tensor_dim0_stride = 1024
  g1[6] = 0;
  g1[7] = 0;
  v4i z4 = {0, 0, 0, 0};
#if defined(__clang_major__) && (__clang_major__ >= 23)
  v8i z8 = {0, 0, 0, 0, 0, 0, 0, 0};
  __builtin_amdgcn_tensor_load_to_lds(g0, g1, z4, z4, z8, 0);
#else
  __builtin_amdgcn_tensor_load_to_lds(g0, g1, z4, z4, 0);
#endif
}
#endif

__global__ __launch_bounds__(32)
void sde_em_kernel(const float* __restrict__ coeffs, const float* __restrict__ times,
                   const float* __restrict__ noise,  const float* __restrict__ W_init,
                   const float* __restrict__ b_init, const float* __restrict__ W_in,
                   const float* __restrict__ b_in,   const float* __restrict__ W_h,
                   const float* __restrict__ b_h,    const float* __restrict__ W_out,
                   const float* __restrict__ b_out,  const float* __restrict__ W_g,
                   const float* __restrict__ b_g,    const float* __restrict__ W_dec,
                   const float* __restrict__ b_dec,  float* __restrict__ out) {
  __shared__ __bf16 wsw[5][2 * 4 * 32 * 16];   // swizzled weights, 40KB
  __shared__ __bf16 bufY[16 * 64];
  __shared__ __bf16 bufH[2][16 * 64];
  // per-column constant table: 0:Win0 1:bin 2:Wg0 3:bg 4:bh0 5:bh1 6:bout 7:Wdec
  __shared__ float  ctab[8][64];
#if USE_TDM
  __shared__ float  bufDW[2][16 * 64];         // TDM double-buffered noise tile
#endif

  const int tid = threadIdx.x;
  const int l15 = tid & 15;
  const int kh  = tid >> 4;
  const int b0  = blockIdx.x * 16;

  // ---- one-time: swizzle weights into B-fragment layout in LDS ----
  {
    const float* srcs[5] = { W_in + 64, W_h, W_h + 64 * 64, W_out, W_g + 64 };
#pragma unroll
    for (int mat = 0; mat < 5; ++mat) {
      const float* s = srcs[mat];
      for (int kt = 0; kt < 2; ++kt)
        for (int nt = 0; nt < 4; ++nt) {
          __bf16* dst = &wsw[mat][((((kt << 2) | nt) * 32) + tid) * 16];
          const int n  = nt * 16 + l15;
          const int kb = kt * 32 + kh * 8;
#pragma unroll
          for (int e = 0; e < 16; ++e)
            dst[e] = (__bf16)s[(kb + e + (e & 8)) * 64 + n];
        }
    }
  }

  // ---- one-time: per-column constants into LDS (keeps them out of VGPRs) ----
#pragma unroll
  for (int c = tid; c < 64; c += 32) {
    ctab[0][c] = W_in[c];        // row 0 of [65,64] (time row)
    ctab[1][c] = b_in[c];
    ctab[2][c] = W_g[c];
    ctab[3][c] = b_g[c];
    ctab[4][c] = b_h[c];
    ctab[5][c] = b_h[64 + c];
    ctab[6][c] = b_out[c];
    ctab[7][c] = W_dec[c];
  }
  __syncthreads();

  const float bdec = b_dec[0];
  const float dt   = times[1] - times[0];
  const float sqdt = sqrtf(dt);

  // ---- y0 = coeffs[:,0,:] @ W_init + b_init, in C-fragment layout ----
  v8f yc[4];
#pragma unroll
  for (int nt = 0; nt < 4; ++nt) {
    const int col = nt * 16 + l15;
#pragma unroll
    for (int r = 0; r < 8; ++r) {
      const int m = r + 8 * kh;
      const float* crow = coeffs + (size_t)(b0 + m) * (Tn * 8);
      float acc = b_init[col];
#pragma unroll
      for (int d = 0; d < 8; ++d) acc = fmaf(crow[d], W_init[d * 64 + col], acc);
      yc[nt][r] = acc;
    }
  }

  // fused decode: out[b, tcur] = y[b,:] . W_dec + b_dec
  auto emit = [&](int tcur) {
    float acc[8];
#pragma unroll
    for (int r = 0; r < 8; ++r) acc[r] = 0.0f;
#pragma unroll
    for (int nt = 0; nt < 4; ++nt) {
      const float wd = ctab[7][nt * 16 + l15];
#pragma unroll
      for (int r = 0; r < 8; ++r) acc[r] = fmaf(yc[nt][r], wd, acc[r]);
    }
#pragma unroll
    for (int msk = 1; msk < 16; msk <<= 1)
#pragma unroll
      for (int r = 0; r < 8; ++r) acc[r] += __shfl_xor(acc[r], msk, 16);
    if (l15 == 0) {
#pragma unroll
      for (int r = 0; r < 8; ++r)
        out[(size_t)(b0 + kh * 8 + r) * Tn + tcur] = acc[r] + bdec;
    }
  };
  emit(0);

#if USE_TDM
  const unsigned lds_dw0 = (unsigned)(uintptr_t)&bufDW[0][0];
  // prologue: kick TDM for step 0
  tdm_load_noise(noise + (size_t)b0 * Hn, lds_dw0);
#endif

  // ---- Euler-Maruyama main loop (serial chain, persistent wave) ----
  for (int i = 0; i < Tn - 1; ++i) {
    const float t = times[i];

    // stage y_old (bf16); bufY stays intact for the whole step
    store_tile(bufY, yc, l15, kh);
    __syncthreads();

    // ---- diffusion FIRST: g = y_old @ Wg' + (t*Wg0 + bg), fold immediately ----
    {
      v8f gc[4];
#pragma unroll
      for (int nt = 0; nt < 4; ++nt) {
        const int col = nt * 16 + l15;
        const float bias = fmaf(t, ctab[2][col], ctab[3][col]);
#pragma unroll
        for (int r = 0; r < 8; ++r) gc[nt][r] = bias;
      }
      gemm16x64(bufY, wsw[4], gc, tid, l15, kh);

#if USE_TDM
      __builtin_amdgcn_s_wait_tensorcnt(0);      // step-i noise tile resident
      if (i + 1 < Tn - 1)                        // pipeline: prefetch step i+1
        tdm_load_noise(noise + (size_t)(i + 1) * (Bn * Hn) + (size_t)b0 * Hn,
                       lds_dw0 + (unsigned)(((i + 1) & 1) * (16 * 64 * 4)));
      const float* dwb = &bufDW[i & 1][0];
#pragma unroll
      for (int nt = 0; nt < 4; ++nt)
#pragma unroll
        for (int r = 0; r < 8; ++r) {
          const float dwv = dwb[(r + 8 * kh) * 64 + nt * 16 + l15];
          yc[nt][r] = fmaf(sqdt * dwv, gc[nt][r], yc[nt][r]);
        }
#else
#pragma unroll
      for (int nt = 0; nt < 4; ++nt)
#pragma unroll
        for (int r = 0; r < 8; ++r) {
          const float dwv = noise[(size_t)i * (Bn * Hn) +
                                  (size_t)(b0 + r + 8 * kh) * Hn + nt * 16 + l15];
          yc[nt][r] = fmaf(sqdt * dwv, gc[nt][r], yc[nt][r]);
        }
#endif
    } // gc + dw dead before the drift chain begins

    // ---- drift chain on y_old (still staged in bufY) ----
    v8f hc[4];
#pragma unroll
    for (int nt = 0; nt < 4; ++nt) {
      const int col = nt * 16 + l15;
      const float bias = fmaf(t, ctab[0][col], ctab[1][col]);
#pragma unroll
      for (int r = 0; r < 8; ++r) hc[nt][r] = bias;
    }
    gemm16x64(bufY, wsw[0], hc, tid, l15, kh);
#pragma unroll
    for (int nt = 0; nt < 4; ++nt)
#pragma unroll
      for (int r = 0; r < 8; ++r) hc[nt][r] = lipswish(hc[nt][r]);
    store_tile(bufH[0], hc, l15, kh);
    __syncthreads();

#pragma unroll
    for (int nt = 0; nt < 4; ++nt) {
      const float bias = ctab[4][nt * 16 + l15];
#pragma unroll
      for (int r = 0; r < 8; ++r) hc[nt][r] = bias;
    }
    gemm16x64(bufH[0], wsw[1], hc, tid, l15, kh);
#pragma unroll
    for (int nt = 0; nt < 4; ++nt)
#pragma unroll
      for (int r = 0; r < 8; ++r) hc[nt][r] = lipswish(hc[nt][r]);
    store_tile(bufH[1], hc, l15, kh);
    __syncthreads();

#pragma unroll
    for (int nt = 0; nt < 4; ++nt) {
      const float bias = ctab[5][nt * 16 + l15];
#pragma unroll
      for (int r = 0; r < 8; ++r) hc[nt][r] = bias;
    }
    gemm16x64(bufH[1], wsw[2], hc, tid, l15, kh);
#pragma unroll
    for (int nt = 0; nt < 4; ++nt)
#pragma unroll
      for (int r = 0; r < 8; ++r) hc[nt][r] = lipswish(hc[nt][r]);
    store_tile(bufH[0], hc, l15, kh);
    __syncthreads();

#pragma unroll
    for (int nt = 0; nt < 4; ++nt) {
      const float bias = ctab[6][nt * 16 + l15];
#pragma unroll
      for (int r = 0; r < 8; ++r) hc[nt][r] = bias;
    }
    gemm16x64(bufH[0], wsw[3], hc, tid, l15, kh);

    // y_{i+1} = (y + diffusion_term) + dt * drift
#pragma unroll
    for (int nt = 0; nt < 4; ++nt)
#pragma unroll
      for (int r = 0; r < 8; ++r)
        yc[nt][r] = fmaf(dt, hc[nt][r], yc[nt][r]);

    emit(i + 1);
    __syncthreads();
  }
}

extern "C" void kernel_launch(void* const* d_in, const int* in_sizes, int n_in,
                              void* d_out, int out_size, void* d_ws, size_t ws_size,
                              hipStream_t stream) {
  (void)in_sizes; (void)n_in; (void)out_size; (void)d_ws; (void)ws_size;
  const float* coeffs = (const float*)d_in[0];
  const float* times  = (const float*)d_in[1];
  const float* noise  = (const float*)d_in[2];
  const float* W_init = (const float*)d_in[3];
  const float* b_init = (const float*)d_in[4];
  const float* W_in   = (const float*)d_in[5];
  const float* b_in   = (const float*)d_in[6];
  const float* W_h    = (const float*)d_in[7];
  const float* b_h    = (const float*)d_in[8];
  const float* W_out  = (const float*)d_in[9];
  const float* b_out  = (const float*)d_in[10];
  const float* W_g    = (const float*)d_in[11];
  const float* b_g    = (const float*)d_in[12];
  const float* W_dec  = (const float*)d_in[13];
  const float* b_dec  = (const float*)d_in[14];
  float* out = (float*)d_out;

  sde_em_kernel<<<dim3(Bn / 16), dim3(32), 0, stream>>>(
      coeffs, times, noise, W_init, b_init, W_in, b_in, W_h, b_h,
      W_out, b_out, W_g, b_g, W_dec, b_dec, out);
}
